// Model_neuMF_790273982929
// MI455X (gfx1250) — matrix operand order repaired and checked by reading, not compiled, not run on hardware
//
#include <hip/hip_runtime.h>
#include <hip/hip_bf16.h>

// ---------------------------------------------------------------------------
// NeuMF forward on MI455X (gfx1250, wave32):
//   gather eU[u], eV[i] -> x[16 x 8] -> relu(x W1^T + b1) -> relu(h1 W2^T + b2)
//   -> h2 W3^T + b3
// Each wave independently processes 16-sample tiles with V_WMMA_F32_16X16X32_F16:
//   L1: K 8->32 pad, N 50->64 (4 tiles)            -> 4 wmma
//   L2: K 50->64 (2 chained K-steps), N 20->32     -> 4 wmma
//   L3: K 20->32, N 1->16 (col 0 valid)            -> 1 wmma
// Biases live in precomputed WMMA C-fragments (VGPR resident across the loop).
// Per layer: issue all WMMAs back-to-back (no D->A/B hazards between them),
// then post-process with packed f16 ops: v_cvt_pk_f16_f32 + v_pk_max_num_f16.
// h1/h2 relayout (C layout -> A layout) goes through per-wave LDS staging;
// wave32 LDS ops are in-order (DScnt), so only a wavefront fence is needed.
// ---------------------------------------------------------------------------

typedef __attribute__((ext_vector_type(16))) _Float16 v16h;
typedef __attribute__((ext_vector_type(8)))  _Float16 v8h;
typedef __attribute__((ext_vector_type(2)))  _Float16 v2h;
typedef __attribute__((ext_vector_type(8)))  float    v8f;
typedef __attribute__((ext_vector_type(4)))  float    v4f;

#define RANK      4
#define F_IN      8     // 2*RANK
#define H1        50
#define H2        20
#define W1_STRIDE 40    // halves per row (32 used + pad), 16B-aligned rows
#define W2_STRIDE 72    // halves per row (64 used + pad), 16B-aligned rows
#define H1_STRIDE 72    // h1 staging: 64 cols used + pad
#define H2_STRIDE 40    // h2 staging: 32 cols used + pad

// A-matrix 16x32 f16 fragment (ISA 7.12.2): lane L (lg=L/16, m=L%16) holds
// two contiguous 8-half chunks: K = kbase+lg*8+[0,8) and kbase+16+lg*8+[0,8).
__device__ __forceinline__ v16h frag_a(const _Float16* row, int lg, int kbase) {
  v8h c0 = *(const v8h*)(row + kbase + lg * 8);
  v8h c1 = *(const v8h*)(row + kbase + 16 + lg * 8);
  v16h r;
#pragma unroll
  for (int j = 0; j < 8; ++j) { r[j] = c0[j]; r[j + 8] = c1[j]; }
  return r;
}

// B-matrix 32x16 f16 fragment (same K order as A): lane L (col n=L%16)
// holds K = kbase+lg*8+[0,8) and kbase+16+lg*8+[0,8).
__device__ __forceinline__ v16h frag_b(const _Float16* row, int lg, int kbase) {
  v8h c0 = *(const v8h*)(row + kbase + lg * 8);
  v8h c1 = *(const v8h*)(row + kbase + 16 + lg * 8);
  v16h r;
#pragma unroll
  for (int j = 0; j < 8; ++j) { r[j] = c0[j]; r[j + 8] = c1[j]; }
  return r;
}

__device__ __forceinline__ v8f wmma_f16(v16h a, v16h b, v8f c) {
  return __builtin_amdgcn_wmma_f32_16x16x32_f16(
      /*neg_a=*/false, a, /*neg_b=*/false, b,
      /*c_mod=*/(short)0, c, /*reuse_a=*/false, /*reuse_b=*/false);
}

__device__ __forceinline__ void wave_fence() {
  // Per-wave LDS store->load ordering: DS ops of one wave are in-order in HW;
  // this just stops the compiler from reordering across the phase boundary.
  __builtin_amdgcn_fence(__ATOMIC_ACQ_REL, "wavefront");
}

// relu + f32->f16 on a C fragment, stored column-wise into staging LDS:
// lane holds column `ncol`, rows r+lg*8, r=0..7. Packed pairs use
// v_cvt_pk_f16_f32 + v_pk_max_num_f16; halves stored b16 / b16_d16_hi.
__device__ __forceinline__ void relu_store_col(_Float16* base, int stride,
                                               int ncol, int lg, v8f acc) {
  const v2h z2 = {};
#pragma unroll
  for (int p = 0; p < 4; ++p) {
    v2h h;
    h[0] = (_Float16)acc[2 * p];
    h[1] = (_Float16)acc[2 * p + 1];
    h = __builtin_elementwise_max(h, z2);
    base[(2 * p + 0 + lg * 8) * stride + ncol] = h[0];
    base[(2 * p + 1 + lg * 8) * stride + ncol] = h[1];
  }
}

__global__ __launch_bounds__(256) void neumf_wmma_kernel(
    const int* __restrict__ users, const int* __restrict__ items,
    const float* __restrict__ eU, const float* __restrict__ eV,
    const float* __restrict__ W1, const float* __restrict__ b1,
    const float* __restrict__ W2, const float* __restrict__ b2,
    const float* __restrict__ W3, const float* __restrict__ b3,
    float* __restrict__ out, int n) {
  __shared__ __align__(16) _Float16 sW1[64 * W1_STRIDE];    // [n64][k32] zero-padded
  __shared__ __align__(16) _Float16 sW2[32 * W2_STRIDE];    // [n32][k64] zero-padded
  __shared__ __align__(16) _Float16 sW3[32];                // [k32] zero-padded
  __shared__ float sB1[64];
  __shared__ float sB2[32];
  __shared__ __align__(16) _Float16 sH1[8][16 * H1_STRIDE]; // per-wave h1 staging
  __shared__ __align__(16) _Float16 sH2[8][16 * H2_STRIDE]; // per-wave h2 staging

  const int tid  = threadIdx.x;
  const int lane = tid & 31;
  const int wave = tid >> 5;
  const int lg   = lane >> 4;  // which half of the wave
  const int ln   = lane & 15;

  // ---- stage weights once per block (f32 -> f16, zero-padded) ----
  for (int i = tid; i < 64 * 32; i += blockDim.x) {
    int r = i >> 5, k = i & 31;
    sW1[r * W1_STRIDE + k] =
        (r < H1 && k < F_IN) ? (_Float16)W1[r * F_IN + k] : (_Float16)0.f;
  }
  for (int i = tid; i < 32 * 64; i += blockDim.x) {
    int r = i >> 6, k = i & 63;
    sW2[r * W2_STRIDE + k] =
        (r < H2 && k < H1) ? (_Float16)W2[r * H1 + k] : (_Float16)0.f;
  }
  if (tid < 32) sW3[tid] = (tid < H2) ? (_Float16)W3[tid] : (_Float16)0.f;
  if (tid < 64) sB1[tid] = (tid < H1) ? b1[tid] : 0.f;
  if (tid < 32) sB2[tid] = (tid < H2) ? b2[tid] : 0.f;
  __syncthreads();  // the only block-wide barrier

  // ---- resident weight B-fragments + bias C-fragments ----
  v16h fB1[4], fB2[2][2], fB3;
  v8f  cB1[4], cB2[2], cB3;
#pragma unroll
  for (int t = 0; t < 4; ++t) {
    fB1[t] = frag_b(&sW1[(t * 16 + ln) * W1_STRIDE], lg, 0);
    const float bv = sB1[t * 16 + ln];        // 0 for cols >= 50
#pragma unroll
    for (int r = 0; r < 8; ++r) cB1[t][r] = bv;
  }
#pragma unroll
  for (int t = 0; t < 2; ++t) {
    const float bv = sB2[t * 16 + ln];        // 0 for cols >= 20
#pragma unroll
    for (int r = 0; r < 8; ++r) cB2[t][r] = bv;
#pragma unroll
    for (int s = 0; s < 2; ++s)
      fB2[t][s] = frag_b(&sW2[(t * 16 + ln) * W2_STRIDE], lg, s * 32);
  }
  {
    v16h z = {};
    fB3 = z;
    if (ln == 0) fB3 = frag_b(sW3, lg, 0);    // only column 0 is live
  }
  {
    const float b3v = b3[0];
#pragma unroll
    for (int r = 0; r < 8; ++r) cB3[r] = b3v;
  }

  // ---- per-wave tile loop: each tile = 16 samples, waves fully decoupled ----
  const int tiles = (n + 15) >> 4;
  const int gw0   = blockIdx.x * 8 + wave;
  const int gws   = gridDim.x * 8;
  for (int tt = gw0; tt < tiles; tt += gws) {
    const int s0 = tt << 4;

    // prefetch next tile's indices into cache (global_prefetch_b8)
    {
      int nt = tt + gws;
      if (nt < tiles && lg == 0) {
        __builtin_prefetch(users + nt * 16 + ln, 0, 0);
        __builtin_prefetch(items + nt * 16 + ln, 0, 0);
      }
    }

    // ---- gather: build layer-1 A fragment directly in registers ----
    // lanes 0-15 hold K=0..7 (= [eu, ev]); all other slots are K-padding (0).
    v16h fA1 = {};
    if (lg == 0) {
      int si = s0 + ln;
      if (si >= n) si = n - 1;
      int u  = users[si];
      int it = items[si];
      v4f eu = *(const v4f*)(eU + (long)u * RANK);
      v4f ev = *(const v4f*)(eV + (long)it * RANK);
#pragma unroll
      for (int j = 0; j < RANK; ++j) {
        fA1[j]        = (_Float16)eu[j];
        fA1[RANK + j] = (_Float16)ev[j];
      }
    }

    // ---- layer 1: all 4 WMMAs first, then packed-f16 relu + stage h1 ----
    v8f acc1[4];
#pragma unroll
    for (int t = 0; t < 4; ++t) acc1[t] = wmma_f16(fA1, fB1[t], cB1[t]);
    _Float16* h1 = &sH1[wave][0];
#pragma unroll
    for (int t = 0; t < 4; ++t)
      relu_store_col(h1, H1_STRIDE, t * 16 + ln, lg, acc1[t]);
    wave_fence();

    // ---- layer 2: 2 tiles x 2 chained K-steps, then relu + stage h2 ----
    const _Float16* arow = &sH1[wave][ln * H1_STRIDE];
    v16h fA2lo = frag_a(arow, lg, 0);
    v16h fA2hi = frag_a(arow, lg, 32);
    v8f acc2[2];
#pragma unroll
    for (int t = 0; t < 2; ++t) {
      acc2[t] = wmma_f16(fA2lo, fB2[t][0], cB2[t]);
      acc2[t] = wmma_f16(fA2hi, fB2[t][1], acc2[t]);
    }
    _Float16* h2 = &sH2[wave][0];
#pragma unroll
    for (int t = 0; t < 2; ++t)
      relu_store_col(h2, H2_STRIDE, t * 16 + ln, lg, acc2[t]);
    wave_fence();

    // ---- layer 3: [16x32] x [32x16], only column 0 real, bias in C ----
    v16h fA3 = frag_a(&sH2[wave][ln * H2_STRIDE], lg, 0);
    v8f acc3 = wmma_f16(fA3, fB3, cB3);

    // C layout: lane 0 holds (N=0, M=0..7), lane 16 holds (N=0, M=8..15).
    if (ln == 0) {
      const int base = s0 + lg * 8;
      if (base + 7 < n) {
        v4f o0 = {acc3[0], acc3[1], acc3[2], acc3[3]};
        v4f o1 = {acc3[4], acc3[5], acc3[6], acc3[7]};
        *(v4f*)(out + base)     = o0;
        *(v4f*)(out + base + 4) = o1;
      } else {
#pragma unroll
        for (int r = 0; r < 8; ++r)
          if (base + r < n) out[base + r] = acc3[r];
      }
    }
    wave_fence();  // protect sH1/sH2 reuse next iteration (WAR)
  }
}

extern "C" void kernel_launch(void* const* d_in, const int* in_sizes, int n_in,
                              void* d_out, int out_size, void* d_ws, size_t ws_size,
                              hipStream_t stream) {
  const int*   users = (const int*)d_in[0];
  const int*   items = (const int*)d_in[1];
  const float* eU    = (const float*)d_in[2];
  const float* eV    = (const float*)d_in[3];
  const float* W1    = (const float*)d_in[4];
  const float* b1    = (const float*)d_in[5];
  const float* W2    = (const float*)d_in[6];
  const float* b2    = (const float*)d_in[7];
  const float* W3    = (const float*)d_in[8];
  const float* b3    = (const float*)d_in[9];
  float* out = (float*)d_out;

  const int n     = in_sizes[0];
  const int tiles = (n + 15) / 16;
  int blocks      = (tiles + 7) / 8;     // 8 waves (tiles) per block-iteration
  if (blocks > 4096) blocks = 4096;      // grid-stride; amortizes weight staging
  if (blocks < 1) blocks = 1;

  neumf_wmma_kernel<<<blocks, 256, 0, stream>>>(users, items, eU, eV, W1, b1,
                                                W2, b2, W3, b3, out, n);
}
